// AttentionModeEncoder_5119601016855
// MI455X (gfx1250) — matrix operand order repaired and checked by reading, not compiled, not run on hardware
//
#include <hip/hip_runtime.h>

// ---------------------------------------------------------------------------
// MHA encoder block for MI455X (gfx1250), wave32, bf16 WMMA with f32 accum.
//   B=4, S=2048, H=16 heads, D=64, HIDDEN=1024.
// Pipeline: cast fp32->bf16 -> fused QKV WMMA GEMM -> flash attention (async
// global->LDS K/V staging, WMMA scores + online softmax + WMMA P*V) ->
// output-projection WMMA GEMM (fp32).
// ---------------------------------------------------------------------------

#define HIDDEN 1024
#define SEQ    2048
#define NB     4
#define NH     16
#define HD     64

#define AS1 __attribute__((address_space(1)))
#define AS3 __attribute__((address_space(3)))

#if defined(__HIP_DEVICE_COMPILE__) &&                                   \
    __has_builtin(__builtin_amdgcn_global_load_async_to_lds_b128) &&     \
    __has_builtin(__builtin_amdgcn_s_wait_asynccnt)
#define USE_ASYNC_LDS 1
#else
#define USE_ASYNC_LDS 0
#endif

typedef __attribute__((ext_vector_type(16))) __bf16 v16bf;
typedef __attribute__((ext_vector_type(8)))  float  v8f;
typedef __attribute__((ext_vector_type(4)))  int    v4i;

union Frag16 { v16bf v; uint4 q[2]; };

// A-matrix 16x32 bf16 fragment (row-major source, leading dim ld).
// ISA layout: lane(0-15) row=lane, K={0..7,16..23}; lane(16-31) K={8..15,24..31}.
__device__ __forceinline__ v16bf load_a_frag(const __bf16* p, int ld, int row0,
                                             int k0, int lane) {
  const int rr = lane & 15, hi = lane >> 4;
  Frag16 f;
  const __bf16* base = p + (size_t)(row0 + rr) * ld + k0 + hi * 8;
  f.q[0] = *reinterpret_cast<const uint4*>(base);
  f.q[1] = *reinterpret_cast<const uint4*>(base + 16);
  return f.v;
}

// B-matrix 32x16 bf16 fragment. Source stored [N][K] row-major (so K is the
// contiguous axis). lane(0-15): col N=lane, K={0..7,16..23}; lane(16-31): K={8..15,24..31}.
__device__ __forceinline__ v16bf load_b_frag(const __bf16* p, int ld, int n0,
                                             int k0, int lane) {
  const int cc = lane & 15, hi = lane >> 4;
  Frag16 f;
  const __bf16* base = p + (size_t)(n0 + cc) * ld + k0 + hi * 8;
  f.q[0] = *reinterpret_cast<const uint4*>(base);
  f.q[1] = *reinterpret_cast<const uint4*>(base + 16);
  return f.v;
}

__device__ __forceinline__ v8f wmma_bf16(v16bf a, v16bf b, v8f c) {
  return __builtin_amdgcn_wmma_f32_16x16x32_bf16(false, a, false, b,
                                                 (short)0, c, false, false);
}

// ---------------------------------------------------------------------------
__global__ void cast_bf16_kernel(const float* __restrict__ src,
                                 __bf16* __restrict__ dst, int n) {
  for (int i = blockIdx.x * blockDim.x + threadIdx.x; i < n;
       i += gridDim.x * blockDim.x)
    dst[i] = (__bf16)src[i];
}

// ---------------------------------------------------------------------------
// Fused QKV projection. grid=(64,16,3), block=128 (4 waves).
// Wave computes a 32x64 output tile. z selects Q/K/V weight+bias+dest.
// Q: scaled by 1/sqrt(64), stored [b,h,s,d]. K: [b,h,s,d]. V: [b,h,d,s] (T).
// ---------------------------------------------------------------------------
__global__ __launch_bounds__(128) void qkv_gemm_kernel(
    const __bf16* __restrict__ xb,
    const __bf16* __restrict__ Wqb, const __bf16* __restrict__ Wkb,
    const __bf16* __restrict__ Wvb,
    const float* __restrict__ bq, const float* __restrict__ bk,
    const float* __restrict__ bv,
    __bf16* __restrict__ Qb, __bf16* __restrict__ Kb,
    __bf16* __restrict__ Vtb) {
  const int lane = threadIdx.x & 31;
  const int wave = threadIdx.x >> 5;
  const int z    = blockIdx.z;
  const __bf16* W    = (z == 0) ? Wqb : (z == 1) ? Wkb : Wvb;
  const float*  bias = (z == 0) ? bq  : (z == 1) ? bk  : bv;
  const int row0 = blockIdx.x * 128 + wave * 32;
  const int n0   = blockIdx.y * 64;

  v8f acc[2][4] = {};
  for (int k = 0; k < HIDDEN; k += 32) {
    __builtin_prefetch(W + (size_t)n0 * HIDDEN + k + 64, 0, 3);
    v16bf a0 = load_a_frag(xb, HIDDEN, row0,      k, lane);
    v16bf a1 = load_a_frag(xb, HIDDEN, row0 + 16, k, lane);
#pragma unroll
    for (int t = 0; t < 4; ++t) {
      v16bf b = load_b_frag(W, HIDDEN, n0 + 16 * t, k, lane);
      acc[0][t] = wmma_bf16(a0, b, acc[0][t]);
      acc[1][t] = wmma_bf16(a1, b, acc[1][t]);
    }
  }

  const int rr = lane & 15, hi = lane >> 4;
#pragma unroll
  for (int t = 0; t < 4; ++t) {
    const int col = n0 + 16 * t + rr;
    const float bs = bias[col];
    const int h = col >> 6, d = col & 63;
#pragma unroll
    for (int mi = 0; mi < 2; ++mi) {
#pragma unroll
      for (int r = 0; r < 8; ++r) {
        const int m  = row0 + 16 * mi + r + 8 * hi;
        const int bb = m >> 11, ss = m & (SEQ - 1);
        const float v = acc[mi][t][r] + bs;
        if (z == 0)
          Qb[(((size_t)(bb * NH + h)) * SEQ + ss) * HD + d] =
              (__bf16)(v * 0.125f);  // fold 1/sqrt(64)
        else if (z == 1)
          Kb[(((size_t)(bb * NH + h)) * SEQ + ss) * HD + d] = (__bf16)v;
        else
          Vtb[(((size_t)(bb * NH + h)) * HD + d) * SEQ + ss] = (__bf16)v;
      }
    }
  }
}

// ---------------------------------------------------------------------------
// Flash attention. grid=(SEQ/64, B*NH), block=128 (4 waves, 16 q-rows each).
// Per 64-key block: async-stage K (8KB) and V^T (8KB) tiles to LDS once per
// block (shared by all 4 waves), then WMMA scores -> mask/clamp -> online
// softmax -> WMMA P*V. Output [b,s,h*64+d] bf16.
// ---------------------------------------------------------------------------
__global__ __launch_bounds__(128) void attention_kernel(
    const __bf16* __restrict__ Qb, const __bf16* __restrict__ Kb,
    const __bf16* __restrict__ Vtb, const int* __restrict__ mask,
    __bf16* __restrict__ attnb) {
  __shared__ __align__(16) float  sS[4][16][64];
  __shared__ __align__(16) __bf16 sP[4][16][64];
  __shared__ __align__(16) __bf16 sK[64][HD];    // [key_local][d]
  __shared__ __align__(16) __bf16 sVt[HD][64];   // [d][key_local]
  __shared__ float sAlpha[4][16];
  __shared__ float sL[4][16];

  const int lane = threadIdx.x & 31;
  const int wave = threadIdx.x >> 5;
  const int rr = lane & 15, hi = lane >> 4;
  const int bh = blockIdx.y;
  const int b  = bh >> 4, h = bh & 15;
  const int q0 = blockIdx.x * 64 + wave * 16;

  const __bf16* Qp  = Qb  + (size_t)bh * SEQ * HD;
  const __bf16* Kp  = Kb  + (size_t)bh * SEQ * HD;
  const __bf16* Vtp = Vtb + (size_t)bh * HD * SEQ;
  const int* mrow = mask + b * SEQ;

  // Q fragments for this wave's 16 rows (d = 0..63 in two 32-wide chunks).
  v16bf Aq[2];
#pragma unroll
  for (int c = 0; c < 2; ++c) Aq[c] = load_a_frag(Qp, HD, q0, 32 * c, lane);

  v8f O[4] = {};
  float m_i = -1e30f, l_i = 0.0f;

  for (int k0 = 0; k0 < SEQ; k0 += 64) {
    // ---- cooperative stage of K tile (contiguous 8KB) and V^T tile -----
    const __bf16* gK = Kp + (size_t)k0 * HD;  // rows k0..k0+63, d contiguous
    const __bf16* gV = Vtp + k0;              // rows d, 64 keys each, stride SEQ
#if USE_ASYNC_LDS
#pragma unroll
    for (int j = 0; j < 4; ++j) {
      const int c = (int)threadIdx.x + 128 * j;  // 16-byte chunk id, 0..511
      __builtin_amdgcn_global_load_async_to_lds_b128(
          (AS1 v4i*)(gK + c * 8), (AS3 v4i*)(&sK[0][0] + c * 8), 0, 0);
      const int dr = c >> 3, off = (c & 7) * 8;
      __builtin_amdgcn_global_load_async_to_lds_b128(
          (AS1 v4i*)(gV + (size_t)dr * SEQ + off),
          (AS3 v4i*)(&sVt[dr][off]), 0, 0);
    }
    __builtin_amdgcn_s_wait_asynccnt(0);
#else
#pragma unroll
    for (int j = 0; j < 4; ++j) {
      const int c = (int)threadIdx.x + 128 * j;
      *reinterpret_cast<uint4*>(&sK[0][0] + c * 8) =
          *reinterpret_cast<const uint4*>(gK + c * 8);
      const int dr = c >> 3, off = (c & 7) * 8;
      *reinterpret_cast<uint4*>(&sVt[dr][off]) =
          *reinterpret_cast<const uint4*>(gV + (size_t)dr * SEQ + off);
    }
#endif
    __syncthreads();

    // ---- scores: 4 key-subtiles x 2 d-chunks of WMMA (K from LDS) ------
#pragma unroll
    for (int t = 0; t < 4; ++t) {
      v8f Sc = {};
#pragma unroll
      for (int c = 0; c < 2; ++c) {
        v16bf bk = load_b_frag(&sK[0][0], HD, 16 * t, 32 * c, lane);
        Sc = wmma_bf16(Aq[c], bk, Sc);
      }
      const int kc = k0 + 16 * t + rr;
      const int mv = mrow[kc];
#pragma unroll
      for (int r = 0; r < 8; ++r) {
        float s = Sc[r];
        s = fminf(fmaxf(s, -50.0f), 50.0f);
        if (mv == 0) s = -50.0f;  // masked_fill(-inf) then clamp => -50
        sS[wave][r + 8 * hi][16 * t + rr] = s;
      }
    }

    // ---- online softmax: lane owns row rr, key-half hi -----------------
    float bm = -1e30f;
#pragma unroll
    for (int j = 0; j < 32; ++j) bm = fmaxf(bm, sS[wave][rr][hi * 32 + j]);
    bm = fmaxf(bm, __shfl_xor(bm, 16, 32));
    const float newm  = fmaxf(m_i, bm);
    const float alpha = __expf(m_i - newm);
    float ps = 0.0f;
#pragma unroll
    for (int j = 0; j < 32; ++j) {
      const float p = __expf(sS[wave][rr][hi * 32 + j] - newm);
      sP[wave][rr][hi * 32 + j] = (__bf16)p;
      ps += p;
    }
    ps += __shfl_xor(ps, 16, 32);
    l_i = l_i * alpha + ps;
    m_i = newm;
    sAlpha[wave][rr] = alpha;  // both halves write identical value

    // ---- rescale running output ---------------------------------------
#pragma unroll
    for (int r = 0; r < 8; ++r) {
      const float a = sAlpha[wave][r + 8 * hi];
#pragma unroll
      for (int t = 0; t < 4; ++t) O[t][r] *= a;
    }

    // ---- O += P(16x64) x V(64x64): 2 key-chunks x 4 d-tiles (V from LDS)
#pragma unroll
    for (int c = 0; c < 2; ++c) {
      Frag16 ap;
      const __bf16* pb = &sP[wave][rr][c * 32 + hi * 8];
      ap.q[0] = *reinterpret_cast<const uint4*>(pb);
      ap.q[1] = *reinterpret_cast<const uint4*>(pb + 16);
#pragma unroll
      for (int t = 0; t < 4; ++t) {
        v16bf bv = load_b_frag(&sVt[0][0], 64, 16 * t, 32 * c, lane);
        O[t] = wmma_bf16(ap.v, bv, O[t]);
      }
    }
    __syncthreads();  // protect sK/sVt before next iteration's staging
  }

  // ---- finalize: divide by l, store [b, q, h*64+d] bf16 -----------------
  sL[wave][rr] = l_i;
#pragma unroll
  for (int t = 0; t < 4; ++t) {
    const int col = h * HD + 16 * t + rr;
#pragma unroll
    for (int r = 0; r < 8; ++r) {
      const int q = q0 + r + 8 * hi;
      const float denom = sL[wave][r + 8 * hi];
      attnb[((size_t)(b * SEQ + q)) * HIDDEN + col] = (__bf16)(O[t][r] / denom);
    }
  }
}

// ---------------------------------------------------------------------------
// Output projection: out = attn @ Wo^T + bo, fp32 result.
// grid=(64,16), block=128 (4 waves), wave tile 32x64.
// ---------------------------------------------------------------------------
__global__ __launch_bounds__(128) void out_gemm_kernel(
    const __bf16* __restrict__ ab, const __bf16* __restrict__ Wob,
    const float* __restrict__ bo, float* __restrict__ out) {
  const int lane = threadIdx.x & 31;
  const int wave = threadIdx.x >> 5;
  const int row0 = blockIdx.x * 128 + wave * 32;
  const int n0   = blockIdx.y * 64;

  v8f acc[2][4] = {};
  for (int k = 0; k < HIDDEN; k += 32) {
    __builtin_prefetch(Wob + (size_t)n0 * HIDDEN + k + 64, 0, 3);
    v16bf a0 = load_a_frag(ab, HIDDEN, row0,      k, lane);
    v16bf a1 = load_a_frag(ab, HIDDEN, row0 + 16, k, lane);
#pragma unroll
    for (int t = 0; t < 4; ++t) {
      v16bf b = load_b_frag(Wob, HIDDEN, n0 + 16 * t, k, lane);
      acc[0][t] = wmma_bf16(a0, b, acc[0][t]);
      acc[1][t] = wmma_bf16(a1, b, acc[1][t]);
    }
  }

  const int rr = lane & 15, hi = lane >> 4;
#pragma unroll
  for (int t = 0; t < 4; ++t) {
    const int col = n0 + 16 * t + rr;
    const float bs = bo[col];
#pragma unroll
    for (int mi = 0; mi < 2; ++mi) {
#pragma unroll
      for (int r = 0; r < 8; ++r) {
        const int m = row0 + 16 * mi + r + 8 * hi;
        out[(size_t)m * HIDDEN + col] = acc[mi][t][r] + bs;
      }
    }
  }
}

// ---------------------------------------------------------------------------
extern "C" void kernel_launch(void* const* d_in, const int* in_sizes, int n_in,
                              void* d_out, int out_size, void* d_ws,
                              size_t ws_size, hipStream_t stream) {
  const float* x    = (const float*)d_in[0];
  const int*   mask = (const int*)d_in[1];
  const float* Wq   = (const float*)d_in[2];
  const float* bq   = (const float*)d_in[3];
  const float* Wk   = (const float*)d_in[4];
  const float* bk   = (const float*)d_in[5];
  const float* Wv   = (const float*)d_in[6];
  const float* bv   = (const float*)d_in[7];
  const float* Wo   = (const float*)d_in[8];
  const float* bo   = (const float*)d_in[9];
  float* out = (float*)d_out;

  // Workspace layout (bytes). attn output reuses the xb region (dead after
  // the QKV GEMM). Total footprint: 72 MB.
  char* ws = (char*)d_ws;
  __bf16* xb  = (__bf16*)(ws + (size_t)0);           // 16 MB [8192,1024]
  __bf16* wqb = (__bf16*)(ws + ((size_t)16 << 20));  //  2 MB [1024,1024]
  __bf16* wkb = (__bf16*)(ws + ((size_t)18 << 20));
  __bf16* wvb = (__bf16*)(ws + ((size_t)20 << 20));
  __bf16* wob = (__bf16*)(ws + ((size_t)22 << 20));
  __bf16* Qb  = (__bf16*)(ws + ((size_t)24 << 20));  // 16 MB [b,h,s,d]
  __bf16* Kb  = (__bf16*)(ws + ((size_t)40 << 20));  // 16 MB [b,h,s,d]
  __bf16* Vtb = (__bf16*)(ws + ((size_t)56 << 20));  // 16 MB [b,h,d,s]
  __bf16* Ab  = xb;                                  // attn out, reuse xb

  const int NX = NB * SEQ * HIDDEN;   // 8,388,608
  const int NW = HIDDEN * HIDDEN;     // 1,048,576
  cast_bf16_kernel<<<2048, 256, 0, stream>>>(x,  xb,  NX);
  cast_bf16_kernel<<<1024, 256, 0, stream>>>(Wq, wqb, NW);
  cast_bf16_kernel<<<1024, 256, 0, stream>>>(Wk, wkb, NW);
  cast_bf16_kernel<<<1024, 256, 0, stream>>>(Wv, wvb, NW);
  cast_bf16_kernel<<<1024, 256, 0, stream>>>(Wo, wob, NW);

  qkv_gemm_kernel<<<dim3(64, 16, 3), 128, 0, stream>>>(
      xb, wqb, wkb, wvb, bq, bk, bv, Qb, Kb, Vtb);

  attention_kernel<<<dim3(SEQ / 64, NB * NH), 128, 0, stream>>>(
      Qb, Kb, Vtb, mask, Ab);

  out_gemm_kernel<<<dim3(64, 16), 128, 0, stream>>>(Ab, wob, bo, out);
}
